// SparseGPMatern52_42537356100201
// MI455X (gfx1250) — matrix operand order repaired and checked by reading, not compiled, not run on hardware
//
#include <hip/hip_runtime.h>
#include <hip/hip_bf16.h>

// Problem sizes (fixed by the reference)
#define MM 1024      // inducing points
#define BB 131072    // query points
#define DD 5         // feature dim
#define ROWS 64      // query rows per block (16 per wave, 4 waves, row-split)
#define KT_STRIDE 1032  // LDS row stride in bf16 elems (1024 + 8 pad -> bank rotation)

typedef __attribute__((ext_vector_type(16))) __bf16    v16bf;
typedef __attribute__((ext_vector_type(8)))  float     v8f;
typedef __attribute__((ext_vector_type(4)))  unsigned  v4u;

union Frag16 { v16bf v; v4u q[2]; };

__device__ __forceinline__ unsigned short f2bf(float f) {
  unsigned u = __float_as_uint(f);
  u += 0x7FFFu + ((u >> 16) & 1u);   // round-to-nearest-even
  return (unsigned short)(u >> 16);
}

// ---------------------------------------------------------------------------
// Kernel 1: params (inv lengthscales, prior variance) + scaled inducing pts Zs
// ---------------------------------------------------------------------------
__global__ void prep_kernel(const float* __restrict__ log_ls,
                            const float* __restrict__ log_var,
                            const float* __restrict__ Z_raw,
                            float* __restrict__ params,
                            float* __restrict__ Zs) {
  const float zsc[5] = {0.15f, 0.1f, 0.05f, 800.0f, 20.0f};
  const float zsh[5] = {0.0f, 0.0f, 0.0f, 600.0f, 5.0f};
  float inv[5];
#pragma unroll
  for (int d = 0; d < DD; ++d) inv[d] = 1.0f / (expf(log_ls[d]) + 1e-8f);

  int idx = blockIdx.x * blockDim.x + threadIdx.x;
  if (idx == 0) {
#pragma unroll
    for (int d = 0; d < DD; ++d) params[d] = inv[d];
    params[5] = expf(log_var[0]);
  }
  if (idx < MM * DD) {
    int d = idx % DD;
    Zs[idx] = (Z_raw[idx] * zsc[d] + zsh[d]) * inv[d];
  }
}

// ---------------------------------------------------------------------------
// Kernel 2: A = matern52(pdist(Zs,Zs)) + jitter*I   (fp32, 1024x1024)
// ---------------------------------------------------------------------------
__global__ void kzz_kernel(const float* __restrict__ Zs,
                           const float* __restrict__ params,
                           float* __restrict__ A) {
  int idx = blockIdx.x * blockDim.x + threadIdx.x;
  if (idx >= MM * MM) return;
  int i = idx >> 10, j = idx & (MM - 1);
  float pv = params[5];
  float d2 = 0.f;
#pragma unroll
  for (int d = 0; d < DD; ++d) { float t = Zs[i*DD+d] - Zs[j*DD+d]; d2 += t*t; }
  d2 = fmaxf(d2, 0.f);
  float dd = sqrtf(d2 + 1e-8f);
  float s = 2.2360679774997896f * dd;
  float k = pv * (1.f + s + 1.6666666666666667f * dd * dd) * expf(-s);
  if (i == j) k += 1e-4f;
  A[idx] = k;
}

// ---------------------------------------------------------------------------
// Kernel 3: in-place Cholesky A = L (lower), single workgroup, 1024 threads.
// Trailing rank-1 update parallelized over the 2-D triangle (32x32 thread
// tiling) -> balanced critical path ~ N^3/6144 instead of N^2/2 per thread.
// ---------------------------------------------------------------------------
__global__ __launch_bounds__(1024) void chol_kernel(float* __restrict__ A) {
  __shared__ float s_inv;
  const int tid = threadIdx.x;
  const int tx = tid & 31;        // column group
  const int ty = tid >> 5;        // row group (0..31)
  for (int k = 0; k < MM; ++k) {
    if (tid == 0) { float dv = sqrtf(A[k*MM+k]); A[k*MM+k] = dv; s_inv = 1.0f / dv; }
    __syncthreads();
    {
      int i = k + 1 + tid;
      if (i < MM) A[i*MM+k] *= s_inv;      // scale column k
    }
    __syncthreads();
    for (int i = k + 1 + ty; i < MM; i += 32) {
      float lik = A[i*MM+k];
      for (int j = k + 1 + tx; j <= i; j += 32)
        A[i*MM+j] = fmaf(-lik, A[j*MM+k], A[i*MM+j]);
    }
    __syncthreads();
  }
}

// ---------------------------------------------------------------------------
// Kernel 4: lower-triangular inverse, one WAVE per column. The forward
// substitution recurrence stays serial in i, but each dot product
// sum_k L[i,k]*y[k] is split across 32 lanes + shfl_xor reduction, with the
// evolving column y kept in LDS (32 columns x 1024 f32 = 128 KB).
// Critical path ~ N^2/64 dot steps instead of N^2/2 serial MACs.
// ---------------------------------------------------------------------------
__global__ __launch_bounds__(1024) void trinv_kernel(const float* __restrict__ A,
                                                     float* __restrict__ Linv) {
  __shared__ float ybuf[32][MM];          // 128 KB: one column per wave
  const int tid  = threadIdx.x;
  const int lane = tid & 31;
  const int w    = tid >> 5;              // wave index 0..31
  const int jc   = blockIdx.x * 32 + w;   // column this wave solves
  float* y = ybuf[w];

  for (int i = jc; i < MM; ++i) {
    float partial = 0.f;
    for (int k = jc + lane; k < i; k += 32)
      partial += A[i*MM + k] * y[k];
#pragma unroll
    for (int off = 16; off; off >>= 1)
      partial += __shfl_xor(partial, off, 32);
    if (lane == 0) {
      float s = ((i == jc) ? 1.0f : 0.0f) - partial;
      y[i] = s / A[i*MM + i];
    }
    __builtin_amdgcn_wave_barrier();      // order lane0's LDS store vs next loads
  }
  // write out the column (strict upper part is zero)
  for (int i = lane; i < MM; i += 32)
    Linv[i*MM + jc] = (i >= jc) ? y[i] : 0.f;
}

// ---------------------------------------------------------------------------
// Kernel 5: fp32 -> bf16 (row-major Linv; B-operand of the big GEMM)
// ---------------------------------------------------------------------------
__global__ void tobf_kernel(const float* __restrict__ Linv,
                            unsigned short* __restrict__ Linvb) {
  int idx = blockIdx.x * blockDim.x + threadIdx.x;
  if (idx < MM * MM) Linvb[idx] = f2bf(Linv[idx]);
}

// ---------------------------------------------------------------------------
// Kernel 6 (main): fused Matern(k_xZ stripe -> LDS) + WMMA GEMM W = k * Linv^T
// and red = rowsum(W^2); out = sqrt(max(pv - red, 1e-6)).
//
// Row-split waves: 128 threads = 4 waves; each wave owns 16 query rows with
// its A-stripe hoisted to VGPRs; all waves sweep the SAME B columns in
// lockstep so B (2 MB, L2-resident) is fetched once per block via WGP$.
// L2 traffic: 2048 blocks x 2 MB = 4 GB (vs 16 GB with n-split waves).
// ---------------------------------------------------------------------------
__global__ __launch_bounds__(128) void var_kernel(const float* __restrict__ x_star,
                                                  const unsigned short* __restrict__ Linvb,
                                                  const float* __restrict__ params,
                                                  const float* __restrict__ Zs,
                                                  float* __restrict__ out) {
  // 64 rows x 1032 bf16 = 132 KB static LDS (CDNA5 WGP has 320 KB)
  __shared__ __attribute__((aligned(16))) unsigned short kTile[ROWS * KT_STRIDE];
  __shared__ float redbuf[4][16];

  const int tid = threadIdx.x;
  const int r0  = blockIdx.x * ROWS;

  float inv[5];
#pragma unroll
  for (int d = 0; d < DD; ++d) inv[d] = params[d];
  const float pv = params[5];

  // Phase 1: compute this block's ROWSx1024 stripe of k_xZ into LDS (bf16)
  for (int idx = tid; idx < ROWS * MM; idx += 128) {
    int r = idx >> 10;
    int j = idx & (MM - 1);
    const float* xr = x_star + (size_t)(r0 + r) * DD;
    float d2 = 0.f;
#pragma unroll
    for (int d = 0; d < DD; ++d) { float t = xr[d] * inv[d] - Zs[j*DD+d]; d2 += t*t; }
    d2 = fmaxf(d2, 0.f);
    float dd = sqrtf(d2 + 1e-8f);
    float s = 2.2360679774997896f * dd;
    float kv = pv * (1.f + s + 1.6666666666666667f * dd * dd) * expf(-s);
    kTile[r * KT_STRIDE + j] = f2bf(kv);
  }
  if (tid < 64) redbuf[tid >> 4][tid & 15] = 0.f;
  __syncthreads();

  // Phase 2: per wave, W(16x1024) = kStripe(16x1024) * Linv^T(1024x1024)
  const int lane = tid & 31;
  const int wave = tid >> 5;        // wave owns rows [wave*16, wave*16+16)
  const int half = lane >> 4;
  const int ln   = lane & 15;

  float rsum[8];
#pragma unroll
  for (int v = 0; v < 8; ++v) rsum[v] = 0.f;

  // A-fragment base in LDS: local row = wave*16 + ln; K-halves offset by 8
  // for upper lanes. Compiler hoists the 32 A-fragments (256 VGPRs) out of
  // the mt loop since they are mt-invariant.
  const unsigned short* arow = kTile + (wave * 16 + ln) * KT_STRIDE + half * 8;

  for (int mt = 0; mt < MM / 16; ++mt) {
    const int mBase = mt * 16;      // identical across all 4 waves -> WGP$ hits
    // B-fragment: W[b][m] = sum_j k[b][j] * Linv[m][j]; B(k,n) = Linv[mBase+n][k]
    const unsigned short* brow = Linvb + (size_t)(mBase + ln) * MM + half * 8;
    v8f c = {};
#pragma unroll
    for (int kk = 0; kk < MM; kk += 32) {
      Frag16 a, b;
      a.q[0] = *(const v4u*)(arow + kk);        // K = kk + h   .. +7
      a.q[1] = *(const v4u*)(arow + kk + 16);   // K = kk+16+h  .. +7
      b.q[0] = *(const v4u*)(brow + kk);        // K = kk + h   .. +7
      b.q[1] = *(const v4u*)(brow + kk + 16);   // K = kk+16+h  .. +7
      c = __builtin_amdgcn_wmma_f32_16x16x32_bf16(
              /*neg_a=*/false, a.v, /*neg_b=*/false, b.v,
              /*c_mod=*/(short)0, c, /*reuse_a=*/false, /*reuse_b=*/false);
    }
#pragma unroll
    for (int v = 0; v < 8; ++v) rsum[v] += c[v] * c[v];
  }

  // D layout: row = v + 8*half, col = ln -> per-row sum via LDS float atomics
  // (each wave reduces into its own 16-entry slice; no cross-wave mixing)
#pragma unroll
  for (int v = 0; v < 8; ++v) atomicAdd(&redbuf[wave][v + half * 8], rsum[v]);
  __syncthreads();

  if (tid < ROWS) {
    // global row = r0 + tid, with tid = wave*16 + localrow
    out[r0 + tid] = sqrtf(fmaxf(pv - redbuf[tid >> 4][tid & 15], 1e-6f));
  }
}

// ---------------------------------------------------------------------------
// Host entry
// ---------------------------------------------------------------------------
extern "C" void kernel_launch(void* const* d_in, const int* in_sizes, int n_in,
                              void* d_out, int out_size, void* d_ws, size_t ws_size,
                              hipStream_t stream) {
  (void)in_sizes; (void)n_in; (void)out_size; (void)ws_size;

  const float* x_star  = (const float*)d_in[0];  // (B,5)
  const float* log_ls  = (const float*)d_in[1];  // (5,)
  const float* log_var = (const float*)d_in[2];  // scalar
  const float* Z_raw   = (const float*)d_in[3];  // (M,5)
  float* out = (float*)d_out;                    // (B,)

  char* ws = (char*)d_ws;
  float* params = (float*)(ws + 0);                                       // 8 f32
  float* Zs     = (float*)(ws + 256);                                     // M*5 f32
  float* A      = (float*)(ws + 32768);                                   // M*M f32 (KZZ -> L)
  float* Linv   = (float*)(ws + 32768 + (size_t)MM * MM * 4);             // M*M f32
  unsigned short* Linvb =
      (unsigned short*)(ws + 32768 + 2 * (size_t)MM * MM * 4);            // M*M bf16
  // total workspace use: ~10.03 MB

  prep_kernel<<<(MM * DD + 255) / 256, 256, 0, stream>>>(log_ls, log_var, Z_raw, params, Zs);
  kzz_kernel<<<(MM * MM) / 256, 256, 0, stream>>>(Zs, params, A);
  chol_kernel<<<1, 1024, 0, stream>>>(A);
  trinv_kernel<<<MM / 32, 1024, 0, stream>>>(A, Linv);
  tobf_kernel<<<(MM * MM) / 256, 256, 0, stream>>>(Linv, Linvb);
  var_kernel<<<BB / ROWS, 128, 0, stream>>>(x_star, Linvb, params, Zs, out);
}
